// MultiHeadAttention_45586782880089
// MI455X (gfx1250) — matrix operand order repaired and checked by reading, not compiled, not run on hardware
//
#include <hip/hip_runtime.h>

// MI455X / gfx1250: wave32, WMMA bf16 16x16x32 f32-acc, async-to-LDS staging.

typedef __bf16 bf16;
typedef __attribute__((ext_vector_type(16))) __bf16 v16bf;
typedef __attribute__((ext_vector_type(8)))  __bf16 v8bf;
typedef __attribute__((ext_vector_type(8)))  float  v8f;

union AB { v16bf v; v8bf h[2]; };
union F8 { v8f v; float f[8]; };

#define NB   2
#define NS   2048
#define ND   1024
#define NH   16
#define NDK  64

// ---------------------------------------------------------------- convert
__global__ void f32_to_bf16_kernel(const float* __restrict__ src,
                                   bf16* __restrict__ dst, int n) {
  int i = blockIdx.x * blockDim.x + threadIdx.x;
  if (i < n) dst[i] = (bf16)src[i];
}

// ---------------------------------------------------------------- GEMM
// A: M x K bf16 row-major. W: N x K bf16 row-major (acts as B^T, so
// out = A @ W^T).  Block = 8 waves owning a 128(M) x 64(N) tile; the 64x32
// weight K-slice is shared by all waves and staged into LDS with
// global_load_async_to_lds_b128 (double-buffered, ASYNCcnt-tracked).
// mode 0: store bf16 remapped (B,H,S,DK)
// mode 1: store bf16 remapped transposed (B,H,DK,S)   (for V)
// mode 2: store f32 row-major + bias                  (final projection)
__global__ __launch_bounds__(256) void gemm_bf16_wmma(
    const bf16* __restrict__ A, const bf16* __restrict__ W,
    void* __restrict__ out, const float* __restrict__ bias,
    int M, int N, int K, int mode) {
  __shared__ __align__(64) bf16 wTile[2][64][32];   // 8 KB double buffer

  const int tid  = threadIdx.x;
  const int lane = tid & 31;
  const int half = lane >> 4;
  const int l16  = lane & 15;
  const int wv   = tid >> 5;

  const int ntn = N >> 6;
  const int bm  = blockIdx.x / ntn;
  const int bn  = blockIdx.x - bm * ntn;
  const int m0  = bm * 128 + wv * 16;
  const int n0  = bn * 64;
  if (m0 >= M) return;

  // staging: each of the 256 threads copies one 16-byte chunk per K-step
  const int sRow = tid >> 2;                    // 0..63  : W row in tile
  const int sChk = tid & 3;                     // 0..3   : 16B chunk of 64B
  const bf16* wSrc = W + (size_t)(n0 + sRow) * K + sChk * 8;
  const unsigned ldsDst[2] = {
      (unsigned)(uintptr_t)&wTile[0][sRow][sChk * 8],
      (unsigned)(uintptr_t)&wTile[1][sRow][sChk * 8]};

  F8 acc[4];
#pragma unroll
  for (int t = 0; t < 4; ++t)
#pragma unroll
    for (int r = 0; r < 8; ++r) acc[t].f[r] = 0.0f;

  // A fragment: lane l16 = row; half selects contiguous K 16-chunk {16h..16h+15}
  const bf16* aRow = A + (size_t)(m0 + l16) * K;

  const int nIter = K >> 5;
  // prime buffer 0 with the k0 = 0 slice
  asm volatile("global_load_async_to_lds_b128 %0, %1, off"
               :: "v"(ldsDst[0]), "v"((unsigned long long)(uintptr_t)wSrc)
               : "memory");

  for (int i = 0; i < nIter; ++i) {
    const int k0  = i << 5;
    const int cur = i & 1;
    if (i + 1 < nIter) {
      const unsigned long long nsrc =
          (unsigned long long)(uintptr_t)(wSrc + k0 + 32);
      asm volatile("global_load_async_to_lds_b128 %0, %1, off"
                   :: "v"(ldsDst[cur ^ 1]), "v"(nsrc) : "memory");
      asm volatile("s_wait_asynccnt 0x1" ::: "memory");
    } else {
      asm volatile("s_wait_asynccnt 0x0" ::: "memory");
    }
    __syncthreads();                   // staged slice visible to all 8 waves

    AB a;
    a.h[0] = *(const v8bf*)(aRow + k0 + half * 16);
    a.h[1] = *(const v8bf*)(aRow + k0 + half * 16 + 8);
    __builtin_prefetch(aRow + k0 + 128, 0, 0);   // global_prefetch_b8
#pragma unroll
    for (int t = 0; t < 4; ++t) {
      // B fragment from LDS: lane l16 = column n; half = contiguous K 16-chunk
      v16bf bfrag = *(const v16bf*)(&wTile[cur][t * 16 + l16][half * 16]);
      acc[t].v = __builtin_amdgcn_wmma_f32_16x16x32_bf16(
          false, a.v, false, bfrag, (short)0, acc[t].v, false, false);
    }
    __syncthreads();                   // done reading before buffer reuse
  }

  // C layout: lane l16 = col, VGPR r -> row = r + 8*half
#pragma unroll
  for (int t = 0; t < 4; ++t) {
#pragma unroll
    for (int r = 0; r < 8; ++r) {
      const int row = m0 + half * 8 + r;
      const int col = n0 + t * 16 + l16;
      const float val = acc[t].f[r];
      if (mode == 2) {
        ((float*)out)[(size_t)row * N + col] = val + bias[col];
      } else {
        const int b = row >> 11, s = row & (NS - 1);
        const int h = col >> 6,  d = col & (NDK - 1);
        bf16* o = (bf16*)out;
        if (mode == 0)
          o[(((size_t)(b * NH + h) * NS + s) * NDK) + d] = (bf16)val;
        else
          o[(((size_t)(b * NH + h) * NDK + d) * NS) + s] = (bf16)val;
      }
    }
  }
}

// ---------------------------------------------------------------- attention
// One wave per (b, h, 16-query tile).  Flash-style online softmax over key
// blocks of 32.  Q,K in (B,H,S,64) bf16; V transposed (B,H,64,S) bf16.
__global__ __launch_bounds__(256) void flash_attn_wmma(
    const bf16* __restrict__ Q, const bf16* __restrict__ Km,
    const bf16* __restrict__ Vt, const int* __restrict__ mask,
    bf16* __restrict__ ctxOut) {
  __shared__ __align__(32) bf16 pLds[8][16][32];   // per-wave P staging (8 KB)

  const int tid  = threadIdx.x;
  const int lane = tid & 31;
  const int half = lane >> 4;
  const int l16  = lane & 15;
  const int wv   = tid >> 5;
  const int wid  = blockIdx.x * 8 + wv;
  const int qt   = wid & 127;          // S/16 = 128 query tiles
  const int h    = (wid >> 7) & (NH - 1);
  const int b    = wid >> 11;
  const int q0   = qt * 16;
  const size_t headQK = (size_t)(b * NH + h) * NS * NDK;

  // Q A-fragments covering dk 0..31 and 32..63
  const bf16* qRow = Q + headQK + (size_t)(q0 + l16) * NDK;
  AB aQ[2];
#pragma unroll
  for (int j = 0; j < 2; ++j) {
    aQ[j].h[0] = *(const v8bf*)(qRow + j * 32 + half * 16);
    aQ[j].h[1] = *(const v8bf*)(qRow + j * 32 + half * 16 + 8);
  }

  F8 cx[4];
  float mI[8], lI[8];
#pragma unroll
  for (int t = 0; t < 4; ++t)
#pragma unroll
    for (int r = 0; r < 8; ++r) cx[t].f[r] = 0.0f;
#pragma unroll
  for (int r = 0; r < 8; ++r) { mI[r] = -3.0e38f; lI[r] = 0.0f; }

  const int kbEnd = q0 + 16;           // causal: keys <= q0+15
  for (int kb = 0; kb < kbEnd; kb += 32) {
    // ---- scores: 2 key tiles x 2 K-steps = 4 WMMAs
    F8 sc[2];
#pragma unroll
    for (int t = 0; t < 2; ++t) {
      const bf16* kRow = Km + headQK + (size_t)(kb + t * 16 + l16) * NDK;
      v16bf bk0 = *(const v16bf*)(kRow + half * 16);
      v16bf bk1 = *(const v16bf*)(kRow + 32 + half * 16);
      v8f z = {};
      z = __builtin_amdgcn_wmma_f32_16x16x32_bf16(false, aQ[0].v, false, bk0,
                                                  (short)0, z, false, false);
      z = __builtin_amdgcn_wmma_f32_16x16x32_bf16(false, aQ[1].v, false, bk1,
                                                  (short)0, z, false, false);
      sc[t].v = z;
    }

    // ---- scale + mask (lane = key column, VGPR r -> query row r+8*half)
    const int key0 = kb + l16;
    const int key1 = kb + 16 + l16;
    const int pad0 = mask[b * NS + key0];
    const int pad1 = mask[b * NS + key1];
    float tmax[8];
#pragma unroll
    for (int r = 0; r < 8; ++r) {
      const int row = q0 + half * 8 + r;
      float v0 = sc[0].f[r] * 0.125f;    // 1/sqrt(64)
      float v1 = sc[1].f[r] * 0.125f;
      if (pad0 || key0 > row) v0 = -1.0e9f;
      if (pad1 || key1 > row) v1 = -1.0e9f;
      sc[0].f[r] = v0; sc[1].f[r] = v1;
      tmax[r] = fmaxf(v0, v1);
    }

    // ---- online softmax: reductions stay inside each 16-lane half,
    // which is exactly one row-group of the WMMA C layout.
#pragma unroll
    for (int r = 0; r < 8; ++r) {
      float t = tmax[r];
      t = fmaxf(t, __shfl_xor(t, 1, 32));
      t = fmaxf(t, __shfl_xor(t, 2, 32));
      t = fmaxf(t, __shfl_xor(t, 4, 32));
      t = fmaxf(t, __shfl_xor(t, 8, 32));
      const float mNew  = fmaxf(mI[r], t);
      const float scale = __expf(mI[r] - mNew);
      mI[r] = mNew;
      const float p0 = __expf(sc[0].f[r] - mNew);
      const float p1 = __expf(sc[1].f[r] - mNew);
      sc[0].f[r] = p0; sc[1].f[r] = p1;
      float rs = p0 + p1;
      rs += __shfl_xor(rs, 1, 32);
      rs += __shfl_xor(rs, 2, 32);
      rs += __shfl_xor(rs, 4, 32);
      rs += __shfl_xor(rs, 8, 32);
      lI[r] = lI[r] * scale + rs;
      cx[0].f[r] *= scale; cx[1].f[r] *= scale;
      cx[2].f[r] *= scale; cx[3].f[r] *= scale;
    }

    // ---- C-layout -> A-layout transpose of P through LDS (intra-wave)
#pragma unroll
    for (int r = 0; r < 8; ++r) {
      const int lr = half * 8 + r;
      pLds[wv][lr][l16]      = (bf16)sc[0].f[r];
      pLds[wv][lr][16 + l16] = (bf16)sc[1].f[r];
    }
    asm volatile("s_wait_dscnt 0" ::: "memory");   // CDNA5 split DS counter
    AB aP;
    aP.h[0] = *(const v8bf*)(&pLds[wv][l16][half * 16]);
    aP.h[1] = *(const v8bf*)(&pLds[wv][l16][half * 16 + 8]);

    // ---- ctx += P @ V  (V transposed -> contiguous B-fragment loads)
#pragma unroll
    for (int t2 = 0; t2 < 4; ++t2) {
      const bf16* vRow =
          Vt + ((size_t)(b * NH + h) * NDK + t2 * 16 + l16) * NS + kb;
      v16bf bv = *(const v16bf*)(vRow + half * 16);
      cx[t2].v = __builtin_amdgcn_wmma_f32_16x16x32_bf16(
          false, aP.v, false, bv, (short)0, cx[t2].v, false, false);
    }
  }

  // ---- epilogue: normalize and store ctx as (B,S,D) bf16
#pragma unroll
  for (int t2 = 0; t2 < 4; ++t2) {
#pragma unroll
    for (int r = 0; r < 8; ++r) {
      const int row = q0 + half * 8 + r;
      const int d   = t2 * 16 + l16;
      const float o = cx[t2].f[r] / lI[r];
      ctxOut[((size_t)(b * NS) + row) * ND + h * NDK + d] = (bf16)o;
    }
  }
}

// ---------------------------------------------------------------- launch
extern "C" void kernel_launch(void* const* d_in, const int* in_sizes, int n_in,
                              void* d_out, int out_size, void* d_ws,
                              size_t ws_size, hipStream_t stream) {
  const float* x   = (const float*)d_in[0];
  const int* mask  = (const int*)d_in[1];
  const float* WQ  = (const float*)d_in[2];
  const float* WK  = (const float*)d_in[3];
  const float* WV  = (const float*)d_in[4];
  const float* WO  = (const float*)d_in[5];
  const float* bO  = (const float*)d_in[6];

  char* ws = (char*)d_ws;
  bf16* xb   = (bf16*)ws;                                  // 8 MB
  bf16* wqb  = (bf16*)(ws + 8ull  * 1024 * 1024);          // 2 MB each
  bf16* wkb  = wqb + (size_t)ND * ND;
  bf16* wvb  = wkb + (size_t)ND * ND;
  bf16* wob  = wvb + (size_t)ND * ND;
  bf16* qB   = (bf16*)(ws + 16ull * 1024 * 1024);          // 8 MB each
  bf16* kB   = qB  + 4ull * 1024 * 1024;
  bf16* vtB  = kB  + 4ull * 1024 * 1024;
  bf16* ctxB = vtB + 4ull * 1024 * 1024;                   // total 48 MB

  const int nx = NB * NS * ND;        // 4,194,304
  const int nw = ND * ND;             // 1,048,576
  f32_to_bf16_kernel<<<(nx + 255) / 256, 256, 0, stream>>>(x,  xb,  nx);
  f32_to_bf16_kernel<<<(nw + 255) / 256, 256, 0, stream>>>(WQ, wqb, nw);
  f32_to_bf16_kernel<<<(nw + 255) / 256, 256, 0, stream>>>(WK, wkb, nw);
  f32_to_bf16_kernel<<<(nw + 255) / 256, 256, 0, stream>>>(WV, wvb, nw);
  f32_to_bf16_kernel<<<(nw + 255) / 256, 256, 0, stream>>>(WO, wob, nw);

  const int M = NB * NS;              // 4096
  const int gemmBlocks = (M / 128) * (ND / 64);      // 512 blocks, 8 waves
  gemm_bf16_wmma<<<gemmBlocks, 256, 0, stream>>>(xb, wqb, qB,  nullptr,
                                                 M, ND, ND, 0);
  gemm_bf16_wmma<<<gemmBlocks, 256, 0, stream>>>(xb, wkb, kB,  nullptr,
                                                 M, ND, ND, 0);
  gemm_bf16_wmma<<<gemmBlocks, 256, 0, stream>>>(xb, wvb, vtB, nullptr,
                                                 M, ND, ND, 1);

  const int attnBlocks = (NB * NH * (NS / 16)) / 8;  // 512
  flash_attn_wmma<<<attnBlocks, 256, 0, stream>>>(qB, kB, vtB, mask, ctxB);

  gemm_bf16_wmma<<<gemmBlocks, 256, 0, stream>>>(ctxB, wob, d_out, bO,
                                                 M, ND, ND, 2);
}
